// Attention_13537736917778
// MI455X (gfx1250) — matrix operand order repaired and checked by reading, not compiled, not run on hardware
//
#include <hip/hip_runtime.h>

// ---- CDNA5 (gfx1250) wave32 WMMA attention -------------------------------
typedef __attribute__((ext_vector_type(16))) _Float16     v16h;
typedef __attribute__((ext_vector_type(8)))  float        v8f;
typedef __attribute__((ext_vector_type(4)))  unsigned int u32x4;

#define DIMC  1024
#define HEADS 16
#define HD    64
#define SEQ   2048
#define BATCH 2
#define MTOT  (BATCH * SEQ)   // 4096 token rows

union Frag { v16h v; u32x4 u[2]; };   // 16 halves = 32B (A or B fragment)
union H8   { u32x4 u; _Float16 h[8]; };

__device__ __forceinline__ v8f wmma_f16(const Frag& a, const Frag& b, v8f c) {
  // D = A(16x32 f16) * B(32x16 f16) + C(16x16 f32)
  return __builtin_amdgcn_wmma_f32_16x16x32_f16(false, a.v, false, b.v,
                                                (short)0, c, false, false);
}

// --------------------------------------------------------------------------
__global__ void cvt_f32_to_f16(const float* __restrict__ in,
                               _Float16* __restrict__ out, int n) {
  int i = blockIdx.x * blockDim.x + threadIdx.x;
  if (i < n) out[i] = (_Float16)in[i];
}

// Build interleaved QKV weight per reference concat+reshape semantics:
// cat channel i (0..3071): source = {Wq,Wk,Wv}[i/1024], row i%1024.
__global__ void build_wqkv(const float* __restrict__ Wq, const float* __restrict__ Wk,
                           const float* __restrict__ Wv, const float* __restrict__ bq,
                           const float* __restrict__ bk, const float* __restrict__ bv,
                           _Float16* __restrict__ Weff, float* __restrict__ beff) {
  int tid = blockIdx.x * blockDim.x + threadIdx.x;   // 3072*1024 threads
  int i = tid >> 10, k = tid & 1023;
  int src = i >> 10, ch = i & 1023;
  const float* W = (src == 0) ? Wq : (src == 1) ? Wk : Wv;
  Weff[tid] = (_Float16)W[(size_t)ch * DIMC + k];
  if (k == 0) {
    const float* bb = (src == 0) ? bq : (src == 1) ? bk : bv;
    beff[i] = bb[ch];
  }
}

// --------------------------------------------------------------------------
// C[M,Ncols] = A[M,K](f16,row-major) * Bw[Ncols,K]^T (f16,row-major) + bias
// Wave computes 32x64; block = 4 waves -> 128x64 tile.
template <typename OutT>
__global__ __launch_bounds__(128) void gemm_wmma(
    const _Float16* __restrict__ A, int lda,
    const _Float16* __restrict__ Bw, int ldb,
    const float* __restrict__ bias,
    OutT* __restrict__ C, int ldc, int K) {
  const int lane = threadIdx.x & 31;
  const int wave = threadIdx.x >> 5;
  const int lm = lane & 15, hi = lane >> 4;
  const int m0 = blockIdx.x * 128 + wave * 32;
  const int c0 = blockIdx.y * 64;

  v8f acc[2][4];
#pragma unroll
  for (int j = 0; j < 4; ++j) {
    float bval = bias[c0 + j * 16 + lm];
#pragma unroll
    for (int i = 0; i < 2; ++i)
#pragma unroll
      for (int e = 0; e < 8; ++e) acc[i][j][e] = bval;
  }

  for (int k0 = 0; k0 < K; k0 += 32) {
    Frag a[2], b[4];
#pragma unroll
    for (int i = 0; i < 2; ++i) {   // A: lane=M, K halves {g*8..}+{16+g*8..}
      const _Float16* p = A + (size_t)(m0 + i * 16 + lm) * lda + k0 + hi * 8;
      a[i].u[0] = *(const u32x4*)p;
      a[i].u[1] = *(const u32x4*)(p + 16);
    }
#pragma unroll
    for (int j = 0; j < 4; ++j) {   // B: lane=N, K halves {g*8..}+{16+g*8..}
      const _Float16* p = Bw + (size_t)(c0 + j * 16 + lm) * ldb + k0 + hi * 8;
      b[j].u[0] = *(const u32x4*)p;
      b[j].u[1] = *(const u32x4*)(p + 16);
    }
#pragma unroll
    for (int i = 0; i < 2; ++i)
#pragma unroll
      for (int j = 0; j < 4; ++j) acc[i][j] = wmma_f16(a[i], b[j], acc[i][j]);
  }

#pragma unroll
  for (int i = 0; i < 2; ++i)
#pragma unroll
    for (int j = 0; j < 4; ++j)
#pragma unroll
      for (int r = 0; r < 8; ++r) {   // C/D: VGPR r -> M=r(+8), lane -> N
        int row = m0 + i * 16 + r + 8 * hi;
        int col = c0 + j * 16 + lm;
        C[(size_t)row * ldc + col] = (OutT)acc[i][j][r];
      }
}

// --------------------------------------------------------------------------
// Per (b,h,n): RMSNorm q,k over head_dim; fold 1/sqrt(64) into q.
// Outputs: qn/kn [bh, n, d]; vT [bh, d, n] (transposed for P*V B-fragments).
__global__ __launch_bounds__(128) void norm_permute(
    const _Float16* __restrict__ cat, const float* __restrict__ qw,
    const float* __restrict__ kw, _Float16* __restrict__ qn,
    _Float16* __restrict__ kn, _Float16* __restrict__ vT) {
  int lane = threadIdx.x & 31;
  int wave = threadIdx.x >> 5;
  int idx = blockIdx.x * 4 + wave;           // b<<15 | h<<11 | n
  int n = idx & 2047;
  int h = (idx >> 11) & 15;
  int b = idx >> 15;
  const _Float16* row = cat + (size_t)(b * SEQ + n) * 3072 + h * 192;
  int d0 = lane * 2;
  float q0 = (float)row[d0],       q1 = (float)row[d0 + 1];
  float k0 = (float)row[64 + d0],  k1 = (float)row[64 + d0 + 1];
  float v0 = (float)row[128 + d0], v1 = (float)row[128 + d0 + 1];
  float sq = q0 * q0 + q1 * q1, sk = k0 * k0 + k1 * k1;
#pragma unroll
  for (int off = 16; off > 0; off >>= 1) {
    sq += __shfl_xor(sq, off);
    sk += __shfl_xor(sk, off);
  }
  float iq = rsqrtf(sq * (1.0f / 64.0f) + 1e-6f) * 0.125f;  // scale folded in
  float ik = rsqrtf(sk * (1.0f / 64.0f) + 1e-6f);
  size_t base = (size_t)idx * 64 + d0;
  qn[base]     = (_Float16)(q0 * iq * qw[d0]);
  qn[base + 1] = (_Float16)(q1 * iq * qw[d0 + 1]);
  kn[base]     = (_Float16)(k0 * ik * kw[d0]);
  kn[base + 1] = (_Float16)(k1 * ik * kw[d0 + 1]);
  size_t vb = ((size_t)(b * HEADS + h) * 64 + d0) * SEQ + n;
  vT[vb]       = (_Float16)v0;
  vT[vb + SEQ] = (_Float16)v1;
}

// --------------------------------------------------------------------------
// One wave per (b, h, 16-query tile). 16x2048 f16 score slab in LDS (64 KB),
// two-pass softmax, then P*V with A-fragments read straight from LDS.
__global__ __launch_bounds__(32) void attention_fa(
    const _Float16* __restrict__ qn, const _Float16* __restrict__ kn,
    const _Float16* __restrict__ vT, _Float16* __restrict__ attn) {
  __shared__ _Float16 S[16 * SEQ];   // exactly 64 KB
  const int lane = threadIdx.x & 31;
  const int lm = lane & 15, hi = lane >> 4;
  const int qt = blockIdx.x & 127;
  const int h  = (blockIdx.x >> 7) & 15;
  const int b  = blockIdx.x >> 11;
  const int bh = b * HEADS + h;
  const int q0 = qt * 16;

  const _Float16* qp = qn + (size_t)bh * SEQ * 64;
  const _Float16* kp = kn + (size_t)bh * SEQ * 64;
  const _Float16* vp = vT + (size_t)bh * 64 * SEQ;

  Frag aq[2];                                   // q rows, d = 0..63
#pragma unroll
  for (int s = 0; s < 2; ++s) {
    const _Float16* p = qp + (size_t)(q0 + lm) * 64 + s * 32 + hi * 8;
    aq[s].u[0] = *(const u32x4*)p;
    aq[s].u[1] = *(const u32x4*)(p + 16);
  }

  // Pass 1: S[q, key] = (q*scale) . k   (scale pre-folded)
  for (int kt = 0; kt < SEQ / 16; ++kt) {
    int key0 = kt * 16;
    Frag bk[2];
#pragma unroll
    for (int s = 0; s < 2; ++s) {               // B: lane=key, K=d {g*8..}+{16+g*8..}
      const _Float16* p = kp + (size_t)(key0 + lm) * 64 + s * 32 + hi * 8;
      bk[s].u[0] = *(const u32x4*)p;
      bk[s].u[1] = *(const u32x4*)(p + 16);
    }
    v8f acc = {};
    acc = wmma_f16(aq[0], bk[0], acc);
    acc = wmma_f16(aq[1], bk[1], acc);
#pragma unroll
    for (int r = 0; r < 8; ++r)
      S[(r + 8 * hi) * SEQ + key0 + lm] = (_Float16)acc[r];
  }
  asm volatile("s_wait_dscnt 0" ::: "memory");

  // Softmax stats: lane handles row lm, half hi of the 2048 keys.
  const int rbase = lm * SEQ + hi * (SEQ / 2);
  float mx = -1e30f;
  for (int j = 0; j < SEQ / 2 / 8; ++j) {
    H8 x; x.u = *(const u32x4*)&S[rbase + j * 8];
#pragma unroll
    for (int e = 0; e < 8; ++e) mx = fmaxf(mx, (float)x.h[e]);
  }
  mx = fmaxf(mx, __shfl_xor(mx, 16));
  float sum = 0.f;
  for (int j = 0; j < SEQ / 2 / 8; ++j) {
    H8 x; x.u = *(const u32x4*)&S[rbase + j * 8];
#pragma unroll
    for (int e = 0; e < 8; ++e) {
      float ev = __expf((float)x.h[e] - mx);
      sum += ev;
      x.h[e] = (_Float16)ev;
    }
    *(u32x4*)&S[rbase + j * 8] = x.u;           // unnormalized probs
  }
  sum += __shfl_xor(sum, 16);
  asm volatile("s_wait_dscnt 0" ::: "memory");

  // Pass 2: O = P * V  (A-fragments of P come straight from LDS)
  v8f o[4] = {};
  for (int k0 = 0; k0 < SEQ; k0 += 32) {
    Frag ap;
    const _Float16* sp = &S[lm * SEQ + k0 + hi * 8];
    ap.u[0] = *(const u32x4*)sp;
    ap.u[1] = *(const u32x4*)(sp + 16);
#pragma unroll
    for (int j = 0; j < 4; ++j) {               // B: lane=d, keys {g*8..}+{16+g*8..} (vT)
      Frag bv;
      const _Float16* p = vp + (size_t)(j * 16 + lm) * SEQ + k0 + hi * 8;
      bv.u[0] = *(const u32x4*)p;
      bv.u[1] = *(const u32x4*)(p + 16);
      o[j] = wmma_f16(ap, bv, o[j]);
    }
  }

#pragma unroll
  for (int r = 0; r < 8; ++r) {
    int m = r + 8 * hi;
    float rs = __shfl(sum, m);                  // lane m holds rowsum[m]
#pragma unroll
    for (int j = 0; j < 4; ++j) {
      float val = o[j][r] / rs;
      attn[(size_t)(b * SEQ + q0 + m) * DIMC + h * 64 + j * 16 + lm] =
          (_Float16)val;
    }
  }
}

// --------------------------------------------------------------------------
extern "C" void kernel_launch(void* const* d_in, const int* in_sizes, int n_in,
                              void* d_out, int out_size, void* d_ws, size_t ws_size,
                              hipStream_t stream) {
  (void)in_sizes; (void)n_in; (void)out_size; (void)ws_size;
  const float* x  = (const float*)d_in[0];
  const float* Wq = (const float*)d_in[1];
  const float* bq = (const float*)d_in[2];
  const float* Wk = (const float*)d_in[3];
  const float* bk = (const float*)d_in[4];
  const float* Wv = (const float*)d_in[5];
  const float* bv = (const float*)d_in[6];
  const float* qw = (const float*)d_in[7];
  const float* kw = (const float*)d_in[8];
  const float* Wo = (const float*)d_in[9];
  const float* bo = (const float*)d_in[10];
  float* out = (float*)d_out;

  char* ws = (char*)d_ws;
  size_t off = 0;
  auto alloc = [&](size_t bytes) -> void* {
    void* p = ws + off;
    off = (off + bytes + 255) & ~(size_t)255;
    return p;
  };
  _Float16* xh   = (_Float16*)alloc((size_t)MTOT * DIMC * 2);
  _Float16* Weff = (_Float16*)alloc((size_t)3072 * DIMC * 2);
  float*    beff = (float*)   alloc((size_t)3072 * 4);
  _Float16* Woh  = (_Float16*)alloc((size_t)DIMC * DIMC * 2);
  _Float16* cat  = (_Float16*)alloc((size_t)MTOT * 3072 * 2);
  _Float16* qnb  = (_Float16*)alloc((size_t)BATCH * HEADS * SEQ * HD * 2);
  _Float16* knb  = (_Float16*)alloc((size_t)BATCH * HEADS * SEQ * HD * 2);
  _Float16* vTb  = (_Float16*)alloc((size_t)BATCH * HEADS * HD * SEQ * 2);
  _Float16* attn = (_Float16*)alloc((size_t)MTOT * DIMC * 2);

  cvt_f32_to_f16<<<(MTOT * DIMC) / 256, 256, 0, stream>>>(x, xh, MTOT * DIMC);
  build_wqkv<<<(3072 * DIMC) / 256, 256, 0, stream>>>(Wq, Wk, Wv, bq, bk, bv,
                                                      Weff, beff);
  cvt_f32_to_f16<<<(DIMC * DIMC) / 256, 256, 0, stream>>>(Wo, Woh, DIMC * DIMC);

  // cat[M,3072] = x @ Weff^T + beff  (channel-interleaved qkv)
  gemm_wmma<_Float16><<<dim3(MTOT / 128, 3072 / 64), 128, 0, stream>>>(
      xh, DIMC, Weff, DIMC, beff, cat, 3072, DIMC);

  norm_permute<<<(BATCH * HEADS * SEQ) / 4, 128, 0, stream>>>(cat, qw, kw, qnb,
                                                              knb, vTb);

  attention_fa<<<BATCH * HEADS * (SEQ / 16), 32, 0, stream>>>(qnb, knb, vTb,
                                                              attn);

  // out[M,1024] = attn @ Wo^T + bo   (f32 output)
  gemm_wmma<float><<<dim3(MTOT / 128, DIMC / 64), 128, 0, stream>>>(
      attn, DIMC, Woh, DIMC, bo, out, DIMC, DIMC);
}
